// TransformerEncoderReadout_70428873720448
// MI455X (gfx1250) — matrix operand order repaired and checked by reading, not compiled, not run on hardware
//
#include <hip/hip_runtime.h>
#include <hip/hip_bf16.h>

typedef __attribute__((ext_vector_type(16))) _Float16 v16h;
typedef __attribute__((ext_vector_type(8)))  _Float16 v8h;
typedef __attribute__((ext_vector_type(4)))  _Float16 v4h;
typedef __attribute__((ext_vector_type(8)))  float    v8f;

#define BMOL 256
#define SEQ  256
#define DIM  64
#define NH   8
#define DHD  8
#define DFF  512
#define MROWS (BMOL*SEQ)   // 65536

// ---------------- WMMA helpers (documented gfx1250 operand layouts) ----------------

// A f16 16x32 (MxK): lanes 0-15: M=lane, K = {0..7,16..23}; lanes 16-31: M=lane-16, K={8..15,24..31}
__device__ __forceinline__ v16h load_a16(const _Float16* base, int ld, int lane) {
  const _Float16* p = base + (lane & 15) * ld + ((lane & 16) ? 8 : 0);
  v8h lo = *(const v8h*)(p);
  v8h hi = *(const v8h*)(p + 16);
  return __builtin_shufflevector(lo, hi, 0,1,2,3,4,5,6,7,8,9,10,11,12,13,14,15);
}

// B f16 32x16 (KxN): lanes 0-15: N=lane, K={0..7,16..23}; lanes 16-31: N=lane-16, K={8..15,24..31}
__device__ __forceinline__ v16h load_b16(const _Float16* base, int ld, int lane) {
  const _Float16* p = base + (lane & 15) * ld + ((lane & 16) ? 8 : 0);
  v8h lo = *(const v8h*)(p);
  v8h hi = *(const v8h*)(p + 16);
  return __builtin_shufflevector(lo, hi, 0,1,2,3,4,5,6,7,8,9,10,11,12,13,14,15);
}

__device__ __forceinline__ v8f wmma16(v16h a, v16h b, v8f c) {
  return __builtin_amdgcn_wmma_f32_16x16x32_f16(false, a, false, b, (short)0, c, false, false);
}

__device__ __forceinline__ float rsum16(float v) {
  v += __shfl_xor(v, 1, 32); v += __shfl_xor(v, 2, 32);
  v += __shfl_xor(v, 4, 32); v += __shfl_xor(v, 8, 32);
  return v;
}
__device__ __forceinline__ float rmax16(float v) {
  v = fmaxf(v, __shfl_xor(v, 1, 32)); v = fmaxf(v, __shfl_xor(v, 2, 32));
  v = fmaxf(v, __shfl_xor(v, 4, 32)); v = fmaxf(v, __shfl_xor(v, 8, 32));
  return v;
}

// ---------------- setup kernels ----------------

__global__ void k_zero4(float4* p, int n) {
  int i = blockIdx.x * blockDim.x + threadIdx.x;
  if (i < n) p[i] = make_float4(0.f, 0.f, 0.f, 0.f);
}

__global__ void k_cvt(const float* s, _Float16* d, int n) {
  int i = blockIdx.x * blockDim.x + threadIdx.x;
  if (i < n) d[i] = (_Float16)s[i];
}

__global__ void k_starts(const int* mol, int* starts, int n) {
  int i = blockIdx.x * blockDim.x + threadIdx.x;
  if (i >= n) return;
  if (i == 0 || mol[i] != mol[i - 1]) starts[mol[i]] = i;
}

__global__ void k_counts(const int* starts, int* counts, int n) {
  int b = threadIdx.x;
  if (b < BMOL) counts[b] = ((b == BMOL - 1) ? n : starts[b + 1]) - starts[b];
}

__global__ void k_scatter(const float* af, const int* mol, const int* starts,
                          float* xf, _Float16* xh, int n) {
  int idx = blockIdx.x * blockDim.x + threadIdx.x;
  if (idx >= n * 16) return;
  int i = idx >> 4, d4 = idx & 15;
  int m = mol[i];
  int row = m * SEQ + (i - starts[m]);
  float4 v = ((const float4*)af)[i * 16 + d4];
  ((float4*)xf)[row * 16 + d4] = v;
  v4h h = { (_Float16)v.x, (_Float16)v.y, (_Float16)v.z, (_Float16)v.w };
  ((v4h*)xh)[row * 16 + d4] = h;
}

// ---------------- QKV projection: [M,64] x [64,192]^T, split into q/k/v head layout ----------------
__global__ __launch_bounds__(256) void k_qkv(const _Float16* xh, const _Float16* wq,
                                             const float* ipb,
                                             _Float16* qh, _Float16* kh, _Float16* vh) {
  int tid = threadIdx.x, lane = tid & 31, w = tid >> 5;
  int gr0 = (blockIdx.x * 8 + w) * 16;
  v16h a0 = load_a16(xh + gr0 * 64,      64, lane);
  v16h a1 = load_a16(xh + gr0 * 64 + 32, 64, lane);
  int colb = lane & 15, rhalf = (lane & 16) ? 8 : 0;
#pragma unroll
  for (int nt = 0; nt < 12; ++nt) {
    v8f acc = {};
    v16h b0 = load_b16(wq + nt * 16 * 64,      64, lane);
    v16h b1 = load_b16(wq + nt * 16 * 64 + 32, 64, lane);
    acc = wmma16(a0, b0, acc);
    acc = wmma16(a1, b1, acc);
    int colg = nt * 16 + colb;
    float bias = ipb[colg];
    int sec = colg >> 6, c64 = colg & 63, hh = c64 >> 3, dh = c64 & 7;
    _Float16* dst = (sec == 0) ? qh : ((sec == 1) ? kh : vh);
    float scale = (sec == 0) ? 0.35355339059327373f : 1.0f;  // 1/sqrt(DH) folded into q
#pragma unroll
    for (int j = 0; j < 8; ++j) {
      int gr = gr0 + rhalf + j;
      int bm = gr >> 8, s = gr & 255;
      dst[(((bm << 3) + hh) * SEQ + s) * DHD + dh] = (_Float16)((acc[j] + bias) * scale);
    }
  }
}

// ---------------- attention: one workgroup per (molecule, head) ----------------
__global__ __launch_bounds__(256) void k_attn(const _Float16* qh, const _Float16* kh,
                                              const _Float16* vh, const int* counts,
                                              _Float16* ao) {
  __shared__ _Float16 ks[SEQ * 32];    // [key][dh padded to 32]
  __shared__ _Float16 vst[16 * SEQ];   // [dh padded to 16][key]
  __shared__ _Float16 qs[8][16 * 32];  // per-wave q strip, K padded
  __shared__ _Float16 pb[8][16 * 32];  // per-wave P chunk (16q x 32k)
  int tid = threadIdx.x, lane = tid & 31, w = tid >> 5;
  int bm = blockIdx.x >> 3, h = blockIdx.x & 7;
  const _Float16* kg = kh + ((bm * NH + h) * SEQ) * DHD;
  const _Float16* vg = vh + ((bm * NH + h) * SEQ) * DHD;
  {
    int key = tid;
#pragma unroll
    for (int d = 0; d < 8; ++d)  ks[key * 32 + d] = kg[key * 8 + d];
#pragma unroll
    for (int d = 8; d < 32; ++d) ks[key * 32 + d] = (_Float16)0.f;
#pragma unroll
    for (int d = 0; d < 8; ++d)  vst[d * SEQ + key] = vg[key * 8 + d];
#pragma unroll
    for (int d = 8; d < 16; ++d) vst[d * SEQ + key] = (_Float16)0.f;
  }
  __syncthreads();
  int cnt = counts[bm];
  int colb = lane & 15, rhalf = (lane & 16) ? 8 : 0;

  for (int it = 0; it < 2; ++it) {
    int qr0 = (w * 2 + it) * 16;
    {  // stage this wave's q strip (per-wave LDS region; same-wave DS ops are in order)
      int r = lane >> 1, c0 = (lane & 1) * 16;
      const _Float16* qrow = qh + ((bm * NH + h) * SEQ + qr0 + r) * DHD;
#pragma unroll
      for (int t = 0; t < 16; ++t) {
        int c = c0 + t;
        qs[w][r * 32 + c] = (c < 8) ? qrow[c] : (_Float16)0.f;
      }
    }
    v16h aq = load_a16(&qs[w][0], 32, lane);
    float sc[16][8];
#pragma unroll
    for (int kt = 0; kt < 16; ++kt) {
      v16h bk = load_b16(&ks[kt * 16 * 32], 32, lane);
      v8f c = {};
      c = wmma16(aq, bk, c);
#pragma unroll
      for (int j = 0; j < 8; ++j) sc[kt][j] = c[j];
    }
    // key-padding mask (count >= 128, so every row has valid keys)
#pragma unroll
    for (int kt = 0; kt < 16; ++kt)
      if (kt * 16 + colb >= cnt) {
#pragma unroll
        for (int j = 0; j < 8; ++j) sc[kt][j] = -1e30f;
      }
    float rl[8];
#pragma unroll
    for (int j = 0; j < 8; ++j) {
      float m = -3.0e38f;
#pragma unroll
      for (int kt = 0; kt < 16; ++kt) m = fmaxf(m, sc[kt][j]);
      m = rmax16(m);
      float l = 0.f;
#pragma unroll
      for (int kt = 0; kt < 16; ++kt) { float e = __expf(sc[kt][j] - m); sc[kt][j] = e; l += e; }
      l = rsum16(l);
      rl[j] = 1.0f / l;
    }
    // o = P @ V via 8 chunks of 32 keys; P transposed C->A layout through per-wave LDS
    v8f acc = {};
#pragma unroll
    for (int kc = 0; kc < 8; ++kc) {
#pragma unroll
      for (int tt = 0; tt < 2; ++tt) {
        int kt = kc * 2 + tt;
#pragma unroll
        for (int j = 0; j < 8; ++j)
          pb[w][(rhalf + j) * 32 + tt * 16 + colb] = (_Float16)sc[kt][j];
      }
      v16h ap = load_a16(&pb[w][0], 32, lane);
      v16h bv = load_b16(&vst[0] + kc * 32, SEQ, lane);
      acc = wmma16(ap, bv, acc);
    }
#pragma unroll
    for (int j = 0; j < 8; ++j) {
      if (colb < 8) {
        int row = qr0 + rhalf + j;
        ao[(bm * SEQ + row) * DIM + h * DHD + colb] = (_Float16)(acc[j] * rl[j]);
      }
    }
  }
}

// ---------------- out-projection + residual + LayerNorm1 ----------------
__global__ __launch_bounds__(256) void k_outproj_ln(const _Float16* ao, const _Float16* wo,
                                                    const float* ob, const float* xf,
                                                    const float* g1, const float* be1,
                                                    float* h1f, _Float16* h1h) {
  int tid = threadIdx.x, lane = tid & 31, w = tid >> 5;
  int gr0 = (blockIdx.x * 8 + w) * 16;
  v16h a0 = load_a16(ao + gr0 * 64,      64, lane);
  v16h a1 = load_a16(ao + gr0 * 64 + 32, 64, lane);
  v8f acc[4] = {};
#pragma unroll
  for (int nt = 0; nt < 4; ++nt) {
    v16h b0 = load_b16(wo + nt * 16 * 64,      64, lane);
    v16h b1 = load_b16(wo + nt * 16 * 64 + 32, 64, lane);
    acc[nt] = wmma16(a0, b0, acc[nt]);
    acc[nt] = wmma16(a1, b1, acc[nt]);
  }
  int colb = lane & 15, rhalf = (lane & 16) ? 8 : 0;
  float bb[4], gg[4], ee[4];
#pragma unroll
  for (int nt = 0; nt < 4; ++nt) {
    int col = nt * 16 + colb;
    bb[nt] = ob[col]; gg[nt] = g1[col]; ee[nt] = be1[col];
  }
#pragma unroll
  for (int j = 0; j < 8; ++j) {
    int row = gr0 + rhalf + j;
    float v[4], s1 = 0.f, s2 = 0.f;
#pragma unroll
    for (int nt = 0; nt < 4; ++nt) {
      int col = nt * 16 + colb;
      float t = acc[nt][j] + bb[nt] + xf[row * 64 + col];
      v[nt] = t; s1 += t; s2 += t * t;
    }
    s1 = rsum16(s1); s2 = rsum16(s2);
    float mean = s1 * (1.f / 64.f);
    float var  = s2 * (1.f / 64.f) - mean * mean;
    float rs = rsqrtf(var + 1e-5f);
#pragma unroll
    for (int nt = 0; nt < 4; ++nt) {
      int col = nt * 16 + colb;
      float wv = (v[nt] - mean) * rs * gg[nt] + ee[nt];
      h1f[row * 64 + col] = wv;
      h1h[row * 64 + col] = (_Float16)wv;
    }
  }
}

// ---------------- fused FFN: relu(h1 @ w1^T + b1) @ w2^T + b2, residual + LayerNorm2 ----------------
// Each wave's 16x512 ReLU strip lives in a wave-private LDS tile (16KB); same-wave DS ops
// are in-order, so no barriers are needed between producing and consuming the strip.
// Saves the 2 x 67MB HBM round-trip of the FFN intermediate.
__global__ __launch_bounds__(128) void k_ffn_fused(const _Float16* h1h, const _Float16* w1h,
                                                   const float* b1, const _Float16* w2h,
                                                   const float* b2, const float* h1f,
                                                   const float* g2, const float* be2,
                                                   float* ln2) {
  __shared__ _Float16 ffbuf[4][16 * DFF];  // 4 waves * 16KB = 64KB
  int tid = threadIdx.x, lane = tid & 31, w = tid >> 5;
  int gr0 = (blockIdx.x * 4 + w) * 16;
  int colb = lane & 15, rhalf = (lane & 16) ? 8 : 0;
  __builtin_prefetch(h1f + gr0 * 64, 0, 0);  // residual rows -> global_prefetch_b8
  _Float16* fb = &ffbuf[w][0];

  // stage 1: ff = relu(h1 @ w1^T + b1), written to wave-private LDS strip
  v16h a0 = load_a16(h1h + gr0 * 64,      64, lane);
  v16h a1 = load_a16(h1h + gr0 * 64 + 32, 64, lane);
#pragma unroll 4
  for (int nt = 0; nt < 32; ++nt) {
    v8f acc = {};
    v16h b0  = load_b16(w1h + nt * 16 * 64,      64, lane);
    v16h b1v = load_b16(w1h + nt * 16 * 64 + 32, 64, lane);
    acc = wmma16(a0, b0, acc);
    acc = wmma16(a1, b1v, acc);
    int col = nt * 16 + colb;
    float bias = b1[col];
#pragma unroll
    for (int j = 0; j < 8; ++j)
      fb[(rhalf + j) * DFF + col] = (_Float16)fmaxf(acc[j] + bias, 0.f);
  }

  // stage 2: y = ff @ w2^T + b2 ; A operands stream from LDS (ds_load_b128)
  v8f acc2[4] = {};
#pragma unroll 4
  for (int kc = 0; kc < 16; ++kc) {
    v16h a = load_a16(fb + kc * 32, DFF, lane);
#pragma unroll
    for (int nt = 0; nt < 4; ++nt) {
      v16h b = load_b16(w2h + nt * 16 * DFF + kc * 32, DFF, lane);
      acc2[nt] = wmma16(a, b, acc2[nt]);
    }
  }

  // epilogue: + b2 + residual, LayerNorm2
  float bb[4], gg[4], ee[4];
#pragma unroll
  for (int nt = 0; nt < 4; ++nt) {
    int col = nt * 16 + colb;
    bb[nt] = b2[col]; gg[nt] = g2[col]; ee[nt] = be2[col];
  }
#pragma unroll
  for (int j = 0; j < 8; ++j) {
    int row = gr0 + rhalf + j;
    float v[4], s1 = 0.f, s2 = 0.f;
#pragma unroll
    for (int nt = 0; nt < 4; ++nt) {
      int col = nt * 16 + colb;
      float t = acc2[nt][j] + bb[nt] + h1f[row * 64 + col];
      v[nt] = t; s1 += t; s2 += t * t;
    }
    s1 = rsum16(s1); s2 = rsum16(s2);
    float mean = s1 * (1.f / 64.f);
    float var  = s2 * (1.f / 64.f) - mean * mean;
    float rs = rsqrtf(var + 1e-5f);
#pragma unroll
    for (int nt = 0; nt < 4; ++nt) {
      int col = nt * 16 + colb;
      ln2[row * 64 + col] = (v[nt] - mean) * rs * gg[nt] + ee[nt];
    }
  }
}

// ---------------- mean pool over S (deterministic, no atomics) ----------------
__global__ void k_pool(const float* ln2, float* out) {
  int idx = blockIdx.x * blockDim.x + threadIdx.x;  // 16384
  int b = idx >> 6, d = idx & 63;
  float s = 0.f;
  for (int si = 0; si < SEQ; ++si) s += ln2[(b * SEQ + si) * DIM + d];
  out[idx] = s * (1.f / (float)SEQ);
}

// ---------------- launcher ----------------
extern "C" void kernel_launch(void* const* d_in, const int* in_sizes, int n_in,
                              void* d_out, int out_size, void* d_ws, size_t ws_size,
                              hipStream_t stream) {
  const float* af   = (const float*)d_in[0];
  const int*   mol  = (const int*)d_in[1];
  const float* ipw  = (const float*)d_in[2];
  const float* ipb  = (const float*)d_in[3];
  const float* outw = (const float*)d_in[4];
  const float* outb = (const float*)d_in[5];
  const float* w1   = (const float*)d_in[6];
  const float* b1   = (const float*)d_in[7];
  const float* w2   = (const float*)d_in[8];
  const float* b2   = (const float*)d_in[9];
  const float* g1   = (const float*)d_in[10];
  const float* be1  = (const float*)d_in[11];
  const float* g2   = (const float*)d_in[12];
  const float* be2  = (const float*)d_in[13];
  float* out = (float*)d_out;
  int N = in_sizes[0] / DIM;

  char* base = (char*)d_ws;
  size_t off = 0;
  auto carve = [&](size_t bytes) -> char* {
    char* p = base + off;
    off = (off + bytes + 255) & ~(size_t)255;
    return p;
  };
  int*       starts = (int*)carve(BMOL * 4);
  int*       counts = (int*)carve(BMOL * 4);
  _Float16*  wqkv_h = (_Float16*)carve(192 * 64 * 2);
  _Float16*  wout_h = (_Float16*)carve(64 * 64 * 2);
  _Float16*  w1_h   = (_Float16*)carve(DFF * 64 * 2);
  _Float16*  w2_h   = (_Float16*)carve(64 * DFF * 2);
  float*     x_f    = (float*)carve((size_t)MROWS * 64 * 4);
  _Float16*  x_h    = (_Float16*)carve((size_t)MROWS * 64 * 2);
  _Float16*  q_h    = (_Float16*)carve((size_t)MROWS * 64 * 2);
  _Float16*  k_h    = (_Float16*)carve((size_t)MROWS * 64 * 2);
  _Float16*  v_h    = (_Float16*)carve((size_t)MROWS * 64 * 2);
  _Float16*  ao_h   = (_Float16*)carve((size_t)MROWS * 64 * 2);
  float*     h1_f   = (float*)carve((size_t)MROWS * 64 * 4);
  _Float16*  h1_h   = (_Float16*)carve((size_t)MROWS * 64 * 2);
  float*     ln2    = x_f;  // x no longer needed after k_outproj_ln

  k_zero4<<<(MROWS * 16 + 255) / 256, 256, 0, stream>>>((float4*)x_f, MROWS * 16);
  k_zero4<<<(MROWS * 8 + 255) / 256, 256, 0, stream>>>((float4*)x_h, MROWS * 8);

  k_cvt<<<(192 * 64 + 255) / 256, 256, 0, stream>>>(ipw, wqkv_h, 192 * 64);
  k_cvt<<<(64 * 64 + 255) / 256, 256, 0, stream>>>(outw, wout_h, 64 * 64);
  k_cvt<<<(DFF * 64 + 255) / 256, 256, 0, stream>>>(w1, w1_h, DFF * 64);
  k_cvt<<<(64 * DFF + 255) / 256, 256, 0, stream>>>(w2, w2_h, 64 * DFF);

  k_starts<<<(N + 255) / 256, 256, 0, stream>>>(mol, starts, N);
  k_counts<<<1, 256, 0, stream>>>(starts, counts, N);
  k_scatter<<<(N * 16 + 255) / 256, 256, 0, stream>>>(af, mol, starts, x_f, x_h, N);

  k_qkv<<<MROWS / 128, 256, 0, stream>>>(x_h, wqkv_h, ipb, q_h, k_h, v_h);
  k_attn<<<BMOL * NH, 256, 0, stream>>>(q_h, k_h, v_h, counts, ao_h);
  k_outproj_ln<<<MROWS / 128, 256, 0, stream>>>(ao_h, wout_h, outb, x_f, g1, be1, h1_f, h1_h);
  k_ffn_fused<<<MROWS / 64, 128, 0, stream>>>(h1_h, w1_h, b1, w2_h, b2, h1_f, g2, be2, ln2);
  k_pool<<<(BMOL * DIM) / 256, 256, 0, stream>>>(ln2, out);
}
